// EncoderBlocks_2456721293699
// MI455X (gfx1250) — compile-verified, matrix operand order checked by reading, not run on hardware
//
#include <hip/hip_runtime.h>
#include <math.h>
#include <stdint.h>

// Problem constants (from reference)
#define B_   4
#define S_   2048
#define D_   1024
#define H_   16
#define DFF_ 4096
#define DK_  64
#define T_   (B_ * S_)   // 8192 tokens

typedef __bf16 bf16_t;
typedef bf16_t v16bf  __attribute__((ext_vector_type(16)));
typedef bf16_t bf16x8 __attribute__((ext_vector_type(8)));
typedef float  v8f    __attribute__((ext_vector_type(8)));

union Frag { v16bf v; bf16x8 h[2]; };

__device__ __forceinline__ v8f v8f_zero() {
  v8f z; for (int i = 0; i < 8; ++i) z[i] = 0.0f; return z;
}

__device__ __forceinline__ v8f wmma_bf16(v16bf a, v16bf b, v8f c) {
  // D = A(16x32 bf16) * B(32x16 bf16) + C(16x16 f32)
  return __builtin_amdgcn_wmma_f32_16x16x32_bf16(
      /*neg_a=*/false, a, /*neg_b=*/false, b,
      /*c_mod=*/(short)0, c, /*reuse_a=*/false, /*reuse_b=*/false);
}

// CDNA5 async copy global->LDS (16B per lane), tracked by ASYNCcnt.
__device__ __forceinline__ void async_copy_b128(unsigned lds_addr,
                                                const void* gaddr) {
  asm volatile("global_load_async_to_lds_b128 %0, %1, off"
               :: "v"(lds_addr), "v"(gaddr)
               : "memory");
}
__device__ __forceinline__ void wait_async0() {
  asm volatile("s_wait_asynccnt 0x0" ::: "memory");
}

__device__ __forceinline__ float gelu_exact(float x) {
  return 0.5f * x * (1.0f + erff(x * 0.70710678118654752f));
}

// ---------------------------------------------------------------------------
// Weight transpose + fp32 -> bf16 convert:  W[K,N] (row-major) -> WT[N,K]
// ---------------------------------------------------------------------------
__global__ __launch_bounds__(256) void wconv_kernel(
    const float* __restrict__ W, bf16_t* __restrict__ WT, int K, int N) {
  __shared__ float tile[32][33];
  const int kb = blockIdx.x * 32;
  const int nb = blockIdx.y * 32;
  const int tx = threadIdx.x;   // 0..31
  const int ty = threadIdx.y;   // 0..7
  for (int i = 0; i < 32; i += 8)
    tile[ty + i][tx] = W[(size_t)(kb + ty + i) * N + nb + tx];
  __syncthreads();
  for (int i = 0; i < 32; i += 8)
    WT[(size_t)(nb + ty + i) * K + kb + tx] = (bf16_t)tile[tx][ty + i];
}

// ---------------------------------------------------------------------------
// LayerNorm (unbiased std, /(std+eps)) -> bf16. One block per token, D=1024.
// ---------------------------------------------------------------------------
__global__ __launch_bounds__(256) void ln_kernel(
    const float* __restrict__ x, const float* __restrict__ alpha,
    const float* __restrict__ beta, bf16_t* __restrict__ out) {
  const int row = blockIdx.x;
  const int tid = threadIdx.x;
  const float* xr = x + (size_t)row * D_;
  float vals[4], s = 0.0f, ss = 0.0f;
  for (int i = 0; i < 4; ++i) {
    float v = xr[tid + i * 256];
    vals[i] = v; s += v; ss += v * v;
  }
  __shared__ float rs[256], rss[256];
  rs[tid] = s; rss[tid] = ss;
  __syncthreads();
  for (int o = 128; o > 0; o >>= 1) {
    if (tid < o) { rs[tid] += rs[tid + o]; rss[tid] += rss[tid + o]; }
    __syncthreads();
  }
  const float mean = rs[0] * (1.0f / D_);
  const float var  = (rss[0] - (float)D_ * mean * mean) * (1.0f / (D_ - 1));
  const float inv  = 1.0f / (sqrtf(fmaxf(var, 0.0f)) + 1e-6f);
  bf16_t* orow = out + (size_t)row * D_;
  for (int i = 0; i < 4; ++i) {
    int c = tid + i * 256;
    orow[c] = (bf16_t)(alpha[c] * (vals[i] - mean) * inv + beta[c]);
  }
}

// ---------------------------------------------------------------------------
// WMMA GEMM: out[M,N] = A_bf16[M,K] @ WT_bf16[N,K]^T + bias (+gelu) (+resid)
// 256 threads = 8 waves; block tile 128x128; wave tile 64x32; K-step 32.
// A panel (128x32) double-buffered in LDS via global_load_async_to_lds_b128:
// the copy for K-step k+1 runs underneath the WMMAs of K-step k (ASYNCcnt).
// ---------------------------------------------------------------------------
template<bool OUT_BF16, bool GELU_ACT, bool RESID>
__global__ __launch_bounds__(256) void gemm_wmma_kernel(
    const bf16_t* __restrict__ A, const bf16_t* __restrict__ WT,
    const float* __restrict__ bias, const float* __restrict__ resid,
    void* __restrict__ outp, int M, int N, int K) {
  __shared__ __align__(16) bf16_t apanel[2][128 * 32];   // 16 KB

  const int tid  = threadIdx.x;
  const int lane = tid & 31;
  const int wave = tid >> 5;
  const int r  = lane & 15;   // row-in-tile (A) / col-in-tile (B,C)
  const int hf = lane >> 4;   // half-wave select
  const int mblk = blockIdx.y * 128;
  const int m0 = mblk + (wave >> 2) * 64;
  const int n0 = blockIdx.x * 128 + (wave & 3) * 32;

  v8f acc[4][2];
  for (int mi = 0; mi < 4; ++mi)
    for (int ni = 0; ni < 2; ++ni) acc[mi][ni] = v8f_zero();

  const bf16_t* brow[2];
  for (int ni = 0; ni < 2; ++ni)
    brow[ni] = WT + (size_t)(n0 + ni * 16 + r) * K + hf * 8;

  const int mloc = (wave >> 2) * 64;   // wave's row offset inside the panel

  // Stage one 128x32 panel (512 x 16B chunks spread over 256 threads)
  auto stage_panel = [&](int buf, int k0) {
    for (int i = tid; i < 512; i += 256) {
      int row = i >> 2;   // 0..127
      int c   = i & 3;    // 16B chunk within the 64B row slice
      const bf16_t* src = A + (size_t)(mblk + row) * K + k0 + c * 8;
      unsigned dst = (unsigned)(uintptr_t)&apanel[buf][row * 32 + c * 8];
      async_copy_b128(dst, (const void*)src);
    }
  };

  // Prologue: stage K-step 0
  stage_panel(0, 0);
  wait_async0();
  __syncthreads();

  for (int k0 = 0, it = 0; k0 < K; k0 += 32, ++it) {
    const int cur = it & 1;
    // Kick off the async copy for the next K-step (buffer last read at it-1,
    // whose readers were fenced by the barrier at the end of it-1).
    if (k0 + 32 < K) stage_panel(cur ^ 1, k0 + 32);

    Frag a[4], bfr[2];
    for (int mi = 0; mi < 4; ++mi) {
      const bf16_t* ap = &apanel[cur][(mloc + mi * 16 + r) * 32 + hf * 8];
      a[mi].h[0] = *(const bf16x8*)(ap);
      a[mi].h[1] = *(const bf16x8*)(ap + 16);
    }
    for (int ni = 0; ni < 2; ++ni) {
      bfr[ni].h[0] = *(const bf16x8*)(brow[ni] + k0);
      bfr[ni].h[1] = *(const bf16x8*)(brow[ni] + k0 + 16);
    }
    if (k0 + 64 < K)   // CDNA5 global_prefetch_b8 for B two K-steps ahead
      __builtin_prefetch((const void*)(brow[0] + k0 + 64), 0, 0);
    for (int mi = 0; mi < 4; ++mi)
      for (int ni = 0; ni < 2; ++ni)
        acc[mi][ni] = wmma_bf16(a[mi].v, bfr[ni].v, acc[mi][ni]);

    // Next buffer fully landed in LDS + all waves done reading current one.
    wait_async0();
    __syncthreads();
  }

  // Epilogue. C layout: lane(0-15)=col r, VGPR e -> row e (+8 for upper half).
  for (int mi = 0; mi < 4; ++mi) {
    for (int ni = 0; ni < 2; ++ni) {
      const int col  = n0 + ni * 16 + r;
      const int rowb = m0 + mi * 16 + hf * 8;
      const float bval = bias[col];
      for (int e = 0; e < 8; ++e) {
        float val = acc[mi][ni][e] + bval;
        if (GELU_ACT) val = gelu_exact(val);
        if (RESID)    val += resid[(size_t)(rowb + e) * N + col];
        if (OUT_BF16)
          ((bf16_t*)outp)[(size_t)(rowb + e) * N + col] = (bf16_t)val;
        else
          ((float*)outp)[(size_t)(rowb + e) * N + col] = val;
      }
    }
  }
}

// ---------------------------------------------------------------------------
// Flash attention, one (b,h) x 64 query rows per block. 128 threads = 4 waves,
// 16 query rows each. Key loop over 32 tiles of 64 keys, online softmax.
// K tile async-staged (double-buffered), V tile transposed through LDS
// (double-buffered), P staged per-wave in LDS for the P@V WMMA.
// One barrier per key tile; next tile's copies run under current compute.
// ---------------------------------------------------------------------------
__global__ __launch_bounds__(128) void flash_attn_kernel(
    const bf16_t* __restrict__ q, const bf16_t* __restrict__ k,
    const bf16_t* __restrict__ v, const int* __restrict__ msk,
    bf16_t* __restrict__ out) {
  __shared__ __align__(16) bf16_t ktile[2][64 * 64];   // [key][d]  16 KB
  __shared__ __align__(16) bf16_t vt[2][DK_ * 64];     // [d][key]  16 KB
  __shared__ __align__(16) bf16_t pbuf[4][16 * 64];    // per-wave P 8 KB

  const int tid  = threadIdx.x;
  const int lane = tid & 31;
  const int wave = tid >> 5;
  const int r    = lane & 15;
  const int hf   = lane >> 4;

  const int bh = blockIdx.y;
  const int b  = bh / H_;
  const int h  = bh % H_;
  const int q0 = blockIdx.x * 64 + wave * 16;

  const size_t headoff = (size_t)h * DK_;
  const bf16_t* qbase = q + (size_t)(b * S_) * D_ + headoff;
  const bf16_t* kbase = k + (size_t)(b * S_) * D_ + headoff;
  const bf16_t* vbase = v + (size_t)(b * S_) * D_ + headoff;

  // Q fragments (A layout), fixed for the whole kernel: contraction d = 0..63
  Frag qa[2];
  {
    const bf16_t* qp = qbase + (size_t)(q0 + r) * D_ + hf * 8;
    qa[0].h[0] = *(const bf16x8*)(qp);
    qa[0].h[1] = *(const bf16x8*)(qp + 16);
    qa[1].h[0] = *(const bf16x8*)(qp + 32);
    qa[1].h[1] = *(const bf16x8*)(qp + 48);
  }

  // Stage key tile j into buffer buf: K rows async (row-major, no transpose
  // needed -- consumed d-contiguous), V transposed via VALU loads/DS stores.
  auto stage_tile = [&](int j, int buf) {
    for (int i = tid; i < 512; i += 128) {
      int kk = i >> 3;   // key row 0..63
      int c  = i & 7;    // 16B chunk (8 bf16) within the 128B row
      const bf16_t* src = kbase + (size_t)(j * 64 + kk) * D_ + c * 8;
      unsigned dst = (unsigned)(uintptr_t)&ktile[buf][kk * 64 + c * 8];
      async_copy_b128(dst, (const void*)src);
    }
    for (int i = tid; i < 64 * DK_; i += 128) {
      int kk = i >> 6;
      int dd = i & 63;
      vt[buf][dd * 64 + kk] = vbase[(size_t)(j * 64 + kk) * D_ + dd];
    }
  };

  float run_max[8], run_sum[8];
  v8f o[4];
  for (int e = 0; e < 8; ++e) { run_max[e] = -3.0e38f; run_sum[e] = 0.0f; }
  for (int dt = 0; dt < 4; ++dt) o[dt] = v8f_zero();

  // Prologue: stage tile 0
  stage_tile(0, 0);
  wait_async0();
  __syncthreads();

  for (int j = 0; j < S_ / 64; ++j) {
    const int cur = j & 1;
    if (j + 1 < S_ / 64) stage_tile(j + 1, cur ^ 1);

    // Scores: S = Q @ K^T  (B fragment lane = key col -> K row, d-contiguous)
    v8f s[4];
    for (int nt = 0; nt < 4; ++nt) s[nt] = v8f_zero();
    for (int nt = 0; nt < 4; ++nt) {
      const bf16_t* kp = &ktile[cur][(nt * 16 + r) * 64 + hf * 8];
      Frag kb0, kb1;
      kb0.h[0] = *(const bf16x8*)(kp);
      kb0.h[1] = *(const bf16x8*)(kp + 16);
      kb1.h[0] = *(const bf16x8*)(kp + 32);
      kb1.h[1] = *(const bf16x8*)(kp + 48);
      s[nt] = wmma_bf16(qa[0].v, kb0.v, s[nt]);
      s[nt] = wmma_bf16(qa[1].v, kb1.v, s[nt]);
    }

    // Scale (1/sqrt(64)) + key mask
    for (int nt = 0; nt < 4; ++nt) {
      int key = j * 64 + nt * 16 + r;
      float mv = (msk[b * S_ + key] == 0) ? -1.0e9f : 0.0f;
      for (int e = 0; e < 8; ++e) s[nt][e] = s[nt][e] * 0.125f + mv;
    }

    // Online softmax. Row stats reduced over the 16-lane half-groups.
    float sc[8];
    for (int e = 0; e < 8; ++e) {
      float tm = fmaxf(fmaxf(s[0][e], s[1][e]), fmaxf(s[2][e], s[3][e]));
      tm = fmaxf(tm, __shfl_xor(tm, 1));
      tm = fmaxf(tm, __shfl_xor(tm, 2));
      tm = fmaxf(tm, __shfl_xor(tm, 4));
      tm = fmaxf(tm, __shfl_xor(tm, 8));
      float nm = fmaxf(run_max[e], tm);
      sc[e] = __expf(run_max[e] - nm);
      run_max[e] = nm;
      float rsum = 0.0f;
      for (int nt = 0; nt < 4; ++nt) {
        float p = __expf(s[nt][e] - nm);
        s[nt][e] = p;
        rsum += p;
      }
      rsum += __shfl_xor(rsum, 1);
      rsum += __shfl_xor(rsum, 2);
      rsum += __shfl_xor(rsum, 4);
      rsum += __shfl_xor(rsum, 8);
      run_sum[e] = run_sum[e] * sc[e] + rsum;
    }
    for (int dt = 0; dt < 4; ++dt)
      for (int e = 0; e < 8; ++e) o[dt][e] *= sc[e];

    // P -> per-wave LDS (row-major [16 q][64 keys]); DS ops in-order per wave.
    for (int nt = 0; nt < 4; ++nt)
      for (int e = 0; e < 8; ++e)
        pbuf[wave][(hf * 8 + e) * 64 + nt * 16 + r] = (bf16_t)s[nt][e];

    // O += P @ V   (A frag from pbuf, B frag from vt, both K-contiguous)
    for (int ks = 0; ks < 2; ++ks) {
      Frag pa;
      const bf16_t* pp = &pbuf[wave][r * 64 + ks * 32 + hf * 8];
      pa.h[0] = *(const bf16x8*)(pp);
      pa.h[1] = *(const bf16x8*)(pp + 16);
      for (int dt = 0; dt < 4; ++dt) {
        Frag vb;
        const bf16_t* vp = &vt[cur][(dt * 16 + r) * 64 + ks * 32 + hf * 8];
        vb.h[0] = *(const bf16x8*)(vp);
        vb.h[1] = *(const bf16x8*)(vp + 16);
        o[dt] = wmma_bf16(pa.v, vb.v, o[dt]);
      }
    }

    // Next tile landed + all waves done with current buffers.
    wait_async0();
    __syncthreads();
  }

  // Normalize and write bf16 attention output (token-major [T, D])
  for (int dt = 0; dt < 4; ++dt) {
    for (int e = 0; e < 8; ++e) {
      int row = q0 + hf * 8 + e;
      int col = dt * 16 + r;
      out[((size_t)(b * S_) + row) * D_ + headoff + col] =
          (bf16_t)(o[dt][e] / run_sum[e]);
    }
  }
}

// ---------------------------------------------------------------------------
// Host-side orchestration
// ---------------------------------------------------------------------------
extern "C" void kernel_launch(void* const* d_in, const int* in_sizes, int n_in,
                              void* d_out, int out_size, void* d_ws, size_t ws_size,
                              hipStream_t stream) {
  (void)in_sizes; (void)n_in; (void)out_size; (void)ws_size;
  const float* x     = (const float*)d_in[0];
  const int*   msk   = (const int*)d_in[1];
  const float* wq_w  = (const float*)d_in[2];
  const float* wq_b  = (const float*)d_in[3];
  const float* wk_w  = (const float*)d_in[4];
  const float* wk_b  = (const float*)d_in[5];
  const float* wv_w  = (const float*)d_in[6];
  const float* wv_b  = (const float*)d_in[7];
  const float* wo_w  = (const float*)d_in[8];
  const float* wo_b  = (const float*)d_in[9];
  const float* ff_w0 = (const float*)d_in[10];
  const float* ff_b0 = (const float*)d_in[11];
  const float* ff_w1 = (const float*)d_in[12];
  const float* ff_b1 = (const float*)d_in[13];
  const float* ln0_a = (const float*)d_in[14];
  const float* ln0_b = (const float*)d_in[15];
  const float* ln1_a = (const float*)d_in[16];
  const float* ln1_b = (const float*)d_in[17];
  float* out = (float*)d_out;

  char* ws = (char*)d_ws;
  size_t off = 0;
  auto alloc = [&](size_t bytes) -> void* {
    void* p = ws + off;
    off += (bytes + 255) & ~(size_t)255;
    return p;
  };
  bf16_t* wtq  = (bf16_t*)alloc((size_t)D_ * D_ * 2);
  bf16_t* wtk  = (bf16_t*)alloc((size_t)D_ * D_ * 2);
  bf16_t* wtv  = (bf16_t*)alloc((size_t)D_ * D_ * 2);
  bf16_t* wto  = (bf16_t*)alloc((size_t)D_ * D_ * 2);
  bf16_t* wtf0 = (bf16_t*)alloc((size_t)DFF_ * D_ * 2);
  bf16_t* wtf1 = (bf16_t*)alloc((size_t)D_ * DFF_ * 2);
  bf16_t* h_bf  = (bf16_t*)alloc((size_t)T_ * D_ * 2);
  bf16_t* q_bf  = (bf16_t*)alloc((size_t)T_ * D_ * 2);
  bf16_t* k_bf  = (bf16_t*)alloc((size_t)T_ * D_ * 2);
  bf16_t* v_bf  = (bf16_t*)alloc((size_t)T_ * D_ * 2);
  bf16_t* ao_bf = (bf16_t*)alloc((size_t)T_ * D_ * 2);
  float*  x1_f  = (float*) alloc((size_t)T_ * D_ * 4);
  bf16_t* h2_bf = (bf16_t*)alloc((size_t)T_ * D_ * 2);
  bf16_t* g_bf  = (bf16_t*)alloc((size_t)T_ * DFF_ * 2);

  const dim3 wb(32, 8);
  // Weight convert/transpose (fp32 [K,N] -> bf16 [N,K])
  wconv_kernel<<<dim3(D_ / 32, D_ / 32), wb, 0, stream>>>(wq_w, wtq, D_, D_);
  wconv_kernel<<<dim3(D_ / 32, D_ / 32), wb, 0, stream>>>(wk_w, wtk, D_, D_);
  wconv_kernel<<<dim3(D_ / 32, D_ / 32), wb, 0, stream>>>(wv_w, wtv, D_, D_);
  wconv_kernel<<<dim3(D_ / 32, D_ / 32), wb, 0, stream>>>(wo_w, wto, D_, D_);
  wconv_kernel<<<dim3(D_ / 32, DFF_ / 32), wb, 0, stream>>>(ff_w0, wtf0, D_, DFF_);
  wconv_kernel<<<dim3(DFF_ / 32, D_ / 32), wb, 0, stream>>>(ff_w1, wtf1, DFF_, D_);

  // h = LN0(x)
  ln_kernel<<<T_, 256, 0, stream>>>(x, ln0_a, ln0_b, h_bf);

  // q, k, v projections
  const dim3 gdp(D_ / 128, T_ / 128);  // (8, 64)
  gemm_wmma_kernel<true, false, false><<<gdp, 256, 0, stream>>>(
      h_bf, wtq, wq_b, nullptr, q_bf, T_, D_, D_);
  gemm_wmma_kernel<true, false, false><<<gdp, 256, 0, stream>>>(
      h_bf, wtk, wk_b, nullptr, k_bf, T_, D_, D_);
  gemm_wmma_kernel<true, false, false><<<gdp, 256, 0, stream>>>(
      h_bf, wtv, wv_b, nullptr, v_bf, T_, D_, D_);

  // attention
  flash_attn_kernel<<<dim3(S_ / 64, B_ * H_), 128, 0, stream>>>(
      q_bf, k_bf, v_bf, msk, ao_bf);

  // x1 = x + attn @ Wo + bo
  gemm_wmma_kernel<false, false, true><<<gdp, 256, 0, stream>>>(
      ao_bf, wto, wo_b, x, x1_f, T_, D_, D_);

  // h2 = LN1(x1)
  ln_kernel<<<T_, 256, 0, stream>>>(x1_f, ln1_a, ln1_b, h2_bf);

  // g = gelu(h2 @ W0 + b0)
  gemm_wmma_kernel<true, true, false><<<dim3(DFF_ / 128, T_ / 128), 256, 0, stream>>>(
      h2_bf, wtf0, ff_b0, nullptr, g_bf, T_, DFF_, D_);

  // out = x1 + g @ W1 + b1
  gemm_wmma_kernel<false, false, true><<<gdp, 256, 0, stream>>>(
      g_bf, wtf1, ff_b1, x1_f, out, T_, D_, DFF_);
}
